// MultiheadSelfAttention_59115929862310
// MI455X (gfx1250) — compile-verified, matrix operand order checked by reading, not run on hardware
//
#include <hip/hip_runtime.h>

typedef __attribute__((ext_vector_type(16))) __bf16 v16bf;
typedef __attribute__((ext_vector_type(8)))  float  v8f;

#define B_  4
#define S_  2048
#define D_  1024
#define H_  16
#define DH_ 64

// ---------------------------------------------------------------------------
// fp32 -> bf16 elementwise conversion (grid-stride)
// ---------------------------------------------------------------------------
__global__ void cvt_f32_bf16(const float* __restrict__ in, __bf16* __restrict__ out, int n) {
  int i = blockIdx.x * blockDim.x + threadIdx.x;
  int stride = gridDim.x * blockDim.x;
  for (; i < n; i += stride) out[i] = (__bf16)in[i];
}

// ---------------------------------------------------------------------------
// C[M,N] = A[M,K] @ W[N,K]^T  via v_wmma_f32_16x16x32_bf16.
// Register-blocked: each wave owns a 32x64 tile (2 M-tiles x 4 N-tiles,
// 8 fp32 accumulators).  Per K-step: 2 A-fragments + 4 B-fragments feed
// 8 WMMAs -> 2.7x operand reuse vs. 1-tile waves.  Block = 8 waves in a
// 2(M) x 4(N) arrangement -> 64x256 block tile.
// A-fragment: lane (r = lane&15) holds row m+r, 16 contiguous bf16 at
//             k0 + (lane>>4)*16.  B-fragment: lane r holds row n+r of W
//             (== column of W^T), same K slicing.
// mode: 0 = bf16 [M,N]          1 = bf16 split-heads [B,H,S,DH]
//       2 = bf16 split-heads, transposed [B,H,DH,S]   3 = fp32 [M,N]
// ---------------------------------------------------------------------------
__global__ __launch_bounds__(256)
void gemm_bf16_wmma(const __bf16* __restrict__ A, const __bf16* __restrict__ W,
                    __bf16* __restrict__ outb, float* __restrict__ outf,
                    int M, int N, int K, int mode) {
  const int tid  = threadIdx.x;
  const int wave = tid >> 5;
  const int lane = tid & 31;
  const int r    = lane & 15;
  const int half = lane >> 4;
  const int wm   = wave >> 2;            // 0..1
  const int wn   = wave & 3;             // 0..3
  const int m0   = blockIdx.x * 64 + wm * 32;
  const int n0   = blockIdx.y * 256 + wn * 64;

  const __bf16* arow0 = A + (size_t)(m0      + r) * K + half * 16;
  const __bf16* arow1 = A + (size_t)(m0 + 16 + r) * K + half * 16;
  const __bf16* wrow0 = W + (size_t)(n0      + r) * K + half * 16;
  const __bf16* wrow1 = W + (size_t)(n0 + 16 + r) * K + half * 16;
  const __bf16* wrow2 = W + (size_t)(n0 + 32 + r) * K + half * 16;
  const __bf16* wrow3 = W + (size_t)(n0 + 48 + r) * K + half * 16;

  v8f acc[2][4];
#pragma unroll
  for (int i = 0; i < 2; ++i)
#pragma unroll
    for (int j = 0; j < 4; ++j) acc[i][j] = (v8f){};

#pragma unroll 2
  for (int k0 = 0; k0 < K; k0 += 32) {
    const v16bf a0 = *(const v16bf*)(arow0 + k0);
    const v16bf a1 = *(const v16bf*)(arow1 + k0);
    const v16bf b0 = *(const v16bf*)(wrow0 + k0);
    const v16bf b1 = *(const v16bf*)(wrow1 + k0);
    const v16bf b2 = *(const v16bf*)(wrow2 + k0);
    const v16bf b3 = *(const v16bf*)(wrow3 + k0);
    __builtin_prefetch(arow0 + k0 + 128, 0, 1);
    __builtin_prefetch(arow1 + k0 + 128, 0, 1);
    acc[0][0] = __builtin_amdgcn_wmma_f32_16x16x32_bf16(false, a0, false, b0, (short)0, acc[0][0], false, false);
    acc[0][1] = __builtin_amdgcn_wmma_f32_16x16x32_bf16(false, a0, false, b1, (short)0, acc[0][1], false, false);
    acc[0][2] = __builtin_amdgcn_wmma_f32_16x16x32_bf16(false, a0, false, b2, (short)0, acc[0][2], false, false);
    acc[0][3] = __builtin_amdgcn_wmma_f32_16x16x32_bf16(false, a0, false, b3, (short)0, acc[0][3], false, false);
    acc[1][0] = __builtin_amdgcn_wmma_f32_16x16x32_bf16(false, a1, false, b0, (short)0, acc[1][0], false, false);
    acc[1][1] = __builtin_amdgcn_wmma_f32_16x16x32_bf16(false, a1, false, b1, (short)0, acc[1][1], false, false);
    acc[1][2] = __builtin_amdgcn_wmma_f32_16x16x32_bf16(false, a1, false, b2, (short)0, acc[1][2], false, false);
    acc[1][3] = __builtin_amdgcn_wmma_f32_16x16x32_bf16(false, a1, false, b3, (short)0, acc[1][3], false, false);
  }

#pragma unroll
  for (int i = 0; i < 2; ++i) {
#pragma unroll
    for (int jt = 0; jt < 4; ++jt) {
#pragma unroll
      for (int j = 0; j < 8; ++j) {
        const int row = m0 + i * 16 + j + 8 * half;  // C layout: lanes 0-15 -> M=j, 16-31 -> M=j+8
        const int col = n0 + jt * 16 + r;
        const float v = acc[i][jt][j];
        if (mode == 0) {
          outb[(size_t)row * N + col] = (__bf16)v;
        } else if (mode == 3) {
          outf[(size_t)row * N + col] = v;
        } else {
          const int bb = row / S_, ss = row % S_;
          const int hh = col / DH_, dd = col % DH_;
          if (mode == 1)
            outb[(((size_t)(bb * H_ + hh)) * S_ + ss) * DH_ + dd] = (__bf16)v;
          else
            outb[(((size_t)(bb * H_ + hh)) * DH_ + dd) * S_ + ss] = (__bf16)v;
        }
      }
    }
  }
}

// ---------------------------------------------------------------------------
// Causal flash attention, one wave per 16-query tile of one (b,h).
// Q,K: [B,H,S,DH] bf16.   Vt: [B,H,DH,S] bf16 (transposed so the P@V
// B-operand columns are contiguous).   attn out: [B,S,D] bf16.
// Key tiles of 32: scores = two 16x16 WMMA outputs (2 K-steps each),
// online softmax with shfl_xor reductions across the 16-lane N group,
// P staged via LDS (C-layout -> A-layout transpose), 4 WMMAs for P@V.
// ---------------------------------------------------------------------------
__global__ __launch_bounds__(32)
void flash_attn_wmma(const __bf16* __restrict__ Q, const __bf16* __restrict__ Km,
                     const __bf16* __restrict__ Vt, __bf16* __restrict__ attn) {
  __shared__ __align__(32) __bf16 Plds[16][32];

  const int lane = threadIdx.x & 31;
  const int r    = lane & 15;
  const int half = lane >> 4;
  const int bh   = blockIdx.y;
  const int b    = bh / H_;
  const int h    = bh % H_;
  const int q0   = blockIdx.x * 16;

  const __bf16* qp = Q  + (size_t)bh * S_ * DH_;
  const __bf16* kp = Km + (size_t)bh * S_ * DH_;
  const __bf16* vp = Vt + (size_t)bh * DH_ * S_;

  // Q A-fragments for K-dim 0..31 and 32..63 (loaded once)
  const v16bf aq0 = *(const v16bf*)(qp + (size_t)(q0 + r) * DH_ +  0 + half * 16);
  const v16bf aq1 = *(const v16bf*)(qp + (size_t)(q0 + r) * DH_ + 32 + half * 16);

  v8f o0 = {}, o1 = {}, o2 = {}, o3 = {};
  float mi[8], li[8];
#pragma unroll
  for (int j = 0; j < 8; ++j) { mi[j] = -__builtin_inff(); li[j] = 0.f; }

  const float scale = 0.125f;                 // 1/sqrt(64)
  const int nkt = (q0 + 16 + 31) / 32;        // causal: keys 0 .. q0+15

  for (int t = 0; t < nkt; ++t) {
    const int kt = t * 32;

    // --- scores: Q(16x64) x K^T(64x32) -> two 16x16 tiles -----------------
    v8f s0 = {}, s1 = {};
#pragma unroll
    for (int d0 = 0; d0 < 64; d0 += 32) {
      const v16bf a  = (d0 == 0) ? aq0 : aq1;
      const v16bf b0 = *(const v16bf*)(kp + (size_t)(kt      + r) * DH_ + d0 + half * 16);
      const v16bf b1 = *(const v16bf*)(kp + (size_t)(kt + 16 + r) * DH_ + d0 + half * 16);
      s0 = __builtin_amdgcn_wmma_f32_16x16x32_bf16(false, a, false, b0, (short)0, s0, false, false);
      s1 = __builtin_amdgcn_wmma_f32_16x16x32_bf16(false, a, false, b1, (short)0, s1, false, false);
    }

    // --- causal mask + online softmax (per-row stats via 16-lane shuffles) -
    float alpha[8];
#pragma unroll
    for (int j = 0; j < 8; ++j) {
      const int qrow = q0 + j + 8 * half;
      float x0 = ((kt      + r) <= qrow) ? s0[j] * scale : -__builtin_inff();
      float x1 = ((kt + 16 + r) <= qrow) ? s1[j] * scale : -__builtin_inff();

      float tmax = fmaxf(x0, x1);
      tmax = fmaxf(tmax, __shfl_xor(tmax, 8, 16));
      tmax = fmaxf(tmax, __shfl_xor(tmax, 4, 16));
      tmax = fmaxf(tmax, __shfl_xor(tmax, 2, 16));
      tmax = fmaxf(tmax, __shfl_xor(tmax, 1, 16));

      const float nm = fmaxf(mi[j], tmax);
      alpha[j] = __expf(mi[j] - nm);
      const float p0 = __expf(x0 - nm);
      const float p1 = __expf(x1 - nm);

      float rs = p0 + p1;
      rs += __shfl_xor(rs, 8, 16);
      rs += __shfl_xor(rs, 4, 16);
      rs += __shfl_xor(rs, 2, 16);
      rs += __shfl_xor(rs, 1, 16);

      li[j] = li[j] * alpha[j] + rs;
      mi[j] = nm;

      // stage P in LDS: row = j + 8*half, cols r and 16+r
      Plds[j + 8 * half][r]      = (__bf16)p0;
      Plds[j + 8 * half][16 + r] = (__bf16)p1;
    }

    // rescale running O by alpha (row-matched: VGPR j <-> row j + 8*half)
#pragma unroll
    for (int j = 0; j < 8; ++j) {
      o0[j] *= alpha[j]; o1[j] *= alpha[j]; o2[j] *= alpha[j]; o3[j] *= alpha[j];
    }

    __syncthreads();
    // P as A-fragment (16x32), V columns are contiguous rows of Vt
    const v16bf pa = *(const v16bf*)(&Plds[r][half * 16]);
    const v16bf bv0 = *(const v16bf*)(vp + (size_t)( 0 + r) * S_ + kt + half * 16);
    const v16bf bv1 = *(const v16bf*)(vp + (size_t)(16 + r) * S_ + kt + half * 16);
    const v16bf bv2 = *(const v16bf*)(vp + (size_t)(32 + r) * S_ + kt + half * 16);
    const v16bf bv3 = *(const v16bf*)(vp + (size_t)(48 + r) * S_ + kt + half * 16);
    o0 = __builtin_amdgcn_wmma_f32_16x16x32_bf16(false, pa, false, bv0, (short)0, o0, false, false);
    o1 = __builtin_amdgcn_wmma_f32_16x16x32_bf16(false, pa, false, bv1, (short)0, o1, false, false);
    o2 = __builtin_amdgcn_wmma_f32_16x16x32_bf16(false, pa, false, bv2, (short)0, o2, false, false);
    o3 = __builtin_amdgcn_wmma_f32_16x16x32_bf16(false, pa, false, bv3, (short)0, o3, false, false);
    __syncthreads();
  }

  // --- finalize: O /= l, store attn[B,S,D] bf16 ---------------------------
#pragma unroll
  for (int j = 0; j < 8; ++j) {
    const float inv = 1.0f / li[j];
    const int row = q0 + j + 8 * half;
    const size_t base = ((size_t)b * S_ + row) * D_ + h * DH_;
    attn[base +  0 + r] = (__bf16)(o0[j] * inv);
    attn[base + 16 + r] = (__bf16)(o1[j] * inv);
    attn[base + 32 + r] = (__bf16)(o2[j] * inv);
    attn[base + 48 + r] = (__bf16)(o3[j] * inv);
  }
}

// ---------------------------------------------------------------------------
// Host-side orchestration
// ---------------------------------------------------------------------------
extern "C" void kernel_launch(void* const* d_in, const int* in_sizes, int n_in,
                              void* d_out, int out_size, void* d_ws, size_t ws_size,
                              hipStream_t stream) {
  const float* x  = (const float*)d_in[0];
  const float* Wq = (const float*)d_in[1];
  const float* Wk = (const float*)d_in[2];
  const float* Wv = (const float*)d_in[3];
  const float* Wo = (const float*)d_in[4];

  const size_t NX = (size_t)B_ * S_ * D_;   // 8,388,608 elems
  const size_t NW = (size_t)D_ * D_;        // 1,048,576 elems

  char* ws = (char*)d_ws;
  __bf16* xb  = (__bf16*)(ws);
  __bf16* wqb = (__bf16*)(ws + NX * 2);
  __bf16* wkb = (__bf16*)(ws + NX * 2 + NW * 2);
  __bf16* wvb = (__bf16*)(ws + NX * 2 + NW * 4);
  __bf16* wob = (__bf16*)(ws + NX * 2 + NW * 6);
  __bf16* qb  = (__bf16*)(ws + NX * 2 + NW * 8);
  __bf16* kb  = (__bf16*)(ws + NX * 4 + NW * 8);
  __bf16* vb  = (__bf16*)(ws + NX * 6 + NW * 8);
  __bf16* ab  = (__bf16*)(ws + NX * 8 + NW * 8);

  cvt_f32_bf16<<<2048, 256, 0, stream>>>(x,  xb,  (int)NX);
  cvt_f32_bf16<<< 512, 256, 0, stream>>>(Wq, wqb, (int)NW);
  cvt_f32_bf16<<< 512, 256, 0, stream>>>(Wk, wkb, (int)NW);
  cvt_f32_bf16<<< 512, 256, 0, stream>>>(Wv, wvb, (int)NW);
  cvt_f32_bf16<<< 512, 256, 0, stream>>>(Wo, wob, (int)NW);

  const dim3 gg(B_ * S_ / 64, D_ / 256);     // (128, 4), 256 threads = 8 waves (2x4)
  gemm_bf16_wmma<<<gg, 256, 0, stream>>>(xb, wqb, qb, nullptr, B_ * S_, D_, D_, 1);
  gemm_bf16_wmma<<<gg, 256, 0, stream>>>(xb, wkb, kb, nullptr, B_ * S_, D_, D_, 1);
  gemm_bf16_wmma<<<gg, 256, 0, stream>>>(xb, wvb, vb, nullptr, B_ * S_, D_, D_, 2);

  const dim3 ga(S_ / 16, B_ * H_);           // (128, 64), 1 wave per block
  flash_attn_wmma<<<ga, 32, 0, stream>>>(qb, kb, vb, ab);

  gemm_bf16_wmma<<<gg, 256, 0, stream>>>(ab, wob, nullptr, (float*)d_out, B_ * S_, D_, D_, 3);
}
